// MultiHeadSelfAttention_88802743812458
// MI455X (gfx1250) — compile-verified, matrix operand order checked by reading, not run on hardware
//
#include <hip/hip_runtime.h>

#define BB  4
#define LL  2048
#define DD  512
#define HH  8
#define DHH 64
#define N3  1536   // 3*H*DH
#define HD  512    // H*DH

typedef __bf16 bf16;
typedef __attribute__((ext_vector_type(16))) __bf16 v16bf;
typedef __attribute__((ext_vector_type(8)))  float  v8f;

__device__ __forceinline__ bf16 f2bf(float x) {
  unsigned u = __builtin_bit_cast(unsigned, x);
  u += 0x7fffu + ((u >> 16) & 1u);                 // round-to-nearest-even
  unsigned short h = (unsigned short)(u >> 16);
  return __builtin_bit_cast(bf16, h);
}

__device__ __forceinline__ v8f wmma_bf16(v16bf a, v16bf b, v8f c) {
  // D(16x16,f32) = A(16x32,bf16) * B(32x16,bf16) + C
  return __builtin_amdgcn_wmma_f32_16x16x32_bf16(false, a, false, b,
                                                 (short)0, c, false, false);
}

// 16x32 bf16 fragment loader (A layout; B uses the same pattern with
// "row" = output column when the operand is stored K-contiguous).
// Per ISA 7.12.2: lane<16 -> k base 0, lane>=16 -> k base 8;
// vector elems 0..7 = k base+0..7, elems 8..15 = k base+16..23.
__device__ __forceinline__ v16bf load_frag(const bf16* tile, int stride) {
  int lane = threadIdx.x & 31;
  const bf16* p = tile + (size_t)(lane & 15) * stride + ((lane & 16) ? 8 : 0);
  v16bf f;
  *(uint4*)&f         = *(const uint4*)p;        // 8 bf16 = 16B
  *(((uint4*)&f) + 1) = *(const uint4*)(p + 16); // 8 bf16 = 16B
  return f;
}

// Same fragment but sourced from fp32 (LDS attn strip), converted on the fly.
__device__ __forceinline__ v16bf load_frag_f32(const float* tile, int stride) {
  int lane = threadIdx.x & 31;
  const float* p = tile + (size_t)(lane & 15) * stride + ((lane & 16) ? 8 : 0);
  v16bf f;
#pragma unroll
  for (int i = 0; i < 8; ++i) { f[i] = f2bf(p[i]); f[i + 8] = f2bf(p[16 + i]); }
  return f;
}

// ---------------- kernel 1: weight transpose + bf16 convert ----------------
__global__ __launch_bounds__(256)
void k_prep(const float* __restrict__ wqkv, const float* __restrict__ fcw,
            bf16* __restrict__ wqkvT, bf16* __restrict__ fcT) {
  int idx = blockIdx.x * 256 + threadIdx.x;
  if (idx < N3 * DD) {
    int n = idx / DD, k = idx % DD;
    wqkvT[idx] = f2bf(wqkv[(size_t)k * N3 + n]);   // [n][k], K contiguous
  } else {
    int j = idx - N3 * DD;
    int n = j / DD, k = j % DD;
    fcT[j] = f2bf(fcw[(size_t)k * DD + n]);
  }
}

// ---------------- kernel 2: LayerNorm -> bf16 ----------------
__global__ __launch_bounds__(256)
void k_ln(const float* __restrict__ x, const float* __restrict__ g,
          const float* __restrict__ be, bf16* __restrict__ xln) {
  __shared__ float rs[256], rq[256];
  int row = blockIdx.x, tid = threadIdx.x;
  const float* xr = x + (size_t)row * DD;
  float a = xr[tid], b2 = xr[tid + 256];
  rs[tid] = a + b2;
  rq[tid] = a * a + b2 * b2;
  for (int s = 128; s > 0; s >>= 1) {
    __syncthreads();
    if (tid < s) { rs[tid] += rs[tid + s]; rq[tid] += rq[tid + s]; }
  }
  __syncthreads();
  float mu  = rs[0] * (1.0f / DD);
  float var = rq[0] * (1.0f / DD) - mu * mu;
  float inv = rsqrtf(var + 1e-6f);
  bf16* o = xln + (size_t)row * DD;
  o[tid]       = f2bf((a  - mu) * inv * g[tid]       + be[tid]);
  o[tid + 256] = f2bf((b2 - mu) * inv * g[tid + 256] + be[tid + 256]);
}

// ---------------- kernel 3: fused QKV GEMM [8192,512]x[512,1536] ----------------
__global__ __launch_bounds__(256)
void k_qkv(const bf16* __restrict__ xln, const bf16* __restrict__ wqkvT,
           const float* __restrict__ bias, bf16* __restrict__ qk,
           bf16* __restrict__ vT) {
  int wid  = blockIdx.x * 8 + (threadIdx.x >> 5);
  int lane = threadIdx.x & 31;
  int mt = wid / (N3 / 16);
  int nt = wid % (N3 / 16);
  int m0 = mt * 16, n0 = nt * 16;
  const bf16* A  = xln   + (size_t)m0 * DD;
  const bf16* Bt = wqkvT + (size_t)n0 * DD;
  v8f c = {};
#pragma unroll 4
  for (int kk = 0; kk < DD; kk += 32)
    c = wmma_bf16(load_frag(A + kk, DD), load_frag(Bt + kk, DD), c);

  int col = n0 + (lane & 15);
  float bv = bias[col];
#pragma unroll
  for (int v = 0; v < 8; ++v) {
    int row = m0 + v + ((lane & 16) ? 8 : 0);
    float val = c[v] + bv;
    if (col < 1024) {
      qk[(size_t)row * 1024 + col] = f2bf(val);            // Q | K row-major
    } else {                                               // V -> [b][h][dh][l]
      int hh = (col - 1024) >> 6, dh = (col - 1024) & 63;
      int bb = row / LL, l = row % LL;
      vT[(((size_t)bb * HH + hh) * DHH + dh) * LL + l] = f2bf(val);
    }
  }
}

// ---- kernel 4: scores (Q.K^T/sqrt(dh)) + mask + softmax + attn@V (fused) ----
// One block per (b, h, 16-row strip). 8 waves:
//   phase 1: WMMA score strip [16][2048] into LDS
//   phase 2: block softmax; write normalized attn to global AND back to LDS
//   phase 3: out_h strip [16][64] = attn_strip @ V_head via WMMA
//            (V_head = 256KB, stays L2-resident across the 128 strips/head,
//             so the 536MB attn matrix is never re-read from HBM)
__global__ __launch_bounds__(256)
void k_attn(const bf16* __restrict__ qk, const unsigned char* __restrict__ mask,
            const bf16* __restrict__ vT, float* __restrict__ attn,
            bf16* __restrict__ outh) {
  extern __shared__ char smem[];
  float* s    = (float*)smem;      // [16][LL]  scores / attn strip
  float* red  = s + 16 * LL;       // [256]     partial reductions
  float* rowv = red + 256;         // [16]      per-row max / inv-sum
  float* pred = rowv + 16;         // [8][256]  attn@V partial tiles

  int bid = blockIdx.x;
  int lt = bid & (LL / 16 - 1);
  int h  = (bid >> 7) & (HH - 1);
  int b  = bid >> 10;
  int l0 = lt * 16;
  int wave = threadIdx.x >> 5, lane = threadIdx.x & 31;

  // ---- phase 1: scores ----
  const bf16* Qb = qk + ((size_t)(b * LL + l0) * 1024) + h * DHH;
  v16bf q0 = load_frag(Qb,      1024);   // k = dh 0..31
  v16bf q1 = load_frag(Qb + 32, 1024);   // k = dh 32..63

  for (int nt = wave; nt < LL / 16; nt += 8) {
    int m0 = nt * 16;
    const bf16* Kb = qk + ((size_t)(b * LL + m0) * 1024) + 512 + h * DHH;
    v8f c = {};
    c = wmma_bf16(q0, load_frag(Kb,      1024), c);
    c = wmma_bf16(q1, load_frag(Kb + 32, 1024), c);
    int col = m0 + (lane & 15);
#pragma unroll
    for (int v = 0; v < 8; ++v) {
      int row = v + ((lane & 16) ? 8 : 0);
      float val = c[v] * 0.125f;                         // 1/sqrt(64)
      if (mask[((size_t)b * LL + l0 + row) * LL + col]) val = -__builtin_inff();
      s[row * LL + col] = val;
    }
  }
  __syncthreads();

  // ---- phase 2: softmax (16 threads per row) ----
  int r = threadIdx.x >> 4, t = threadIdx.x & 15;
  float m = -__builtin_inff();
  for (int cix = t; cix < LL; cix += 16) m = fmaxf(m, s[r * LL + cix]);
  red[threadIdx.x] = m;
  __syncthreads();
  if (t == 0) {
    float mm = red[r * 16];
    for (int i = 1; i < 16; ++i) mm = fmaxf(mm, red[r * 16 + i]);
    rowv[r] = mm;
  }
  __syncthreads();
  float rmax = rowv[r];
  float ssum = 0.f;
  for (int cix = t; cix < LL; cix += 16) {
    float e = expf(s[r * LL + cix] - rmax);
    s[r * LL + cix] = e;
    ssum += e;
  }
  red[threadIdx.x] = ssum;
  __syncthreads();
  if (t == 0) {
    float sm = 0.f;
    for (int i = 0; i < 16; ++i) sm += red[r * 16 + i];
    rowv[r] = 1.0f / sm;
  }
  __syncthreads();
  float inv = rowv[r];
  float* arow = attn + ((size_t)(h * BB + b) * LL + l0 + r) * LL;  // [h][b][l][m]
  for (int cix = t; cix < LL; cix += 16) {
    float a = s[r * LL + cix] * inv;
    s[r * LL + cix] = a;       // keep normalized strip in LDS for attn@V
    arow[cix] = a;             // required output (only HBM write of attn)
  }
  __syncthreads();

  // ---- phase 3: out_h strip = attn_strip @ V_head ----
  // wave w: d-tile (w&3), k-half (w>>2) of 1024
  int dt = wave & 3, khalf = wave >> 2;
  const bf16*  Bb   = vT + (((size_t)b * HH + h) * DHH + dt * 16) * LL + khalf * 1024;
  const float* Asrc = s + khalf * 1024;
  v8f c2 = {};
#pragma unroll 4
  for (int kk = 0; kk < 1024; kk += 32)
    c2 = wmma_bf16(load_frag_f32(Asrc + kk, LL), load_frag(Bb + kk, LL), c2);
#pragma unroll
  for (int v = 0; v < 8; ++v) {
    int row = v + ((lane & 16) ? 8 : 0);
    pred[wave * 256 + row * 16 + (lane & 15)] = c2[v];
  }
  __syncthreads();
  for (int e = threadIdx.x; e < 16 * DHH; e += 256) {
    int row = e >> 6;                  // 0..15
    int d   = e & 63;                  // 0..63
    int tt  = d >> 4, cc = d & 15;
    float val = pred[tt * 256 + row * 16 + cc] + pred[(tt + 4) * 256 + row * 16 + cc];
    outh[((size_t)b * LL + l0 + row) * HD + h * DHH + d] = f2bf(val);
  }
}

// ---------------- kernel 5: out = out_h @ fc_w + fc_b + residual ----------------
__global__ __launch_bounds__(256)
void k_fc(const bf16* __restrict__ outh, const bf16* __restrict__ fcT,
          const float* __restrict__ bias, const float* __restrict__ resid,
          float* __restrict__ out) {
  int wid  = blockIdx.x * 8 + (threadIdx.x >> 5);
  int lane = threadIdx.x & 31;
  int nt = wid & 31;
  int mt = wid >> 5;
  int m0 = mt * 16, n0 = nt * 16;
  const bf16* A  = outh + (size_t)m0 * HD;
  const bf16* Bt = fcT  + (size_t)n0 * DD;
  v8f c = {};
#pragma unroll 4
  for (int kk = 0; kk < HD; kk += 32)
    c = wmma_bf16(load_frag(A + kk, HD), load_frag(Bt + kk, DD), c);

  int col = n0 + (lane & 15);
  float bv = bias[col];
#pragma unroll
  for (int v = 0; v < 8; ++v) {
    int row = m0 + v + ((lane & 16) ? 8 : 0);
    out[(size_t)row * DD + col] = c[v] + bv + resid[(size_t)row * DD + col];
  }
}

extern "C" void kernel_launch(void* const* d_in, const int* in_sizes, int n_in,
                              void* d_out, int out_size, void* d_ws, size_t ws_size,
                              hipStream_t stream) {
  const float*         input = (const float*)d_in[0];
  const unsigned char* mask  = (const unsigned char*)d_in[1];
  const float*         g     = (const float*)d_in[2];
  const float*         be    = (const float*)d_in[3];
  const float*         wqkv  = (const float*)d_in[4];
  const float*         bqkv  = (const float*)d_in[5];
  const float*         fcw   = (const float*)d_in[6];
  const float*         fcb   = (const float*)d_in[7];

  float* out  = (float*)d_out;
  float* attn = out + (size_t)BB * LL * DD;   // [H*B, L, L] fp32 output region

  // workspace carve (all chunks 256B-multiple => alignment preserved), ~44 MB
  char* w = (char*)d_ws;
  bf16* wqkvT = (bf16*)w; w += (size_t)N3 * DD * 2;
  bf16* fcT   = (bf16*)w; w += (size_t)DD * DD * 2;
  bf16* xln   = (bf16*)w; w += (size_t)BB * LL * DD * 2;
  bf16* qk    = (bf16*)w; w += (size_t)BB * LL * 1024 * 2;
  bf16* vT    = (bf16*)w; w += (size_t)BB * HH * DHH * LL * 2;
  bf16* outh  = (bf16*)w;

  k_prep<<<(N3 * DD + DD * DD) / 256, 256, 0, stream>>>(wqkv, fcw, wqkvT, fcT);
  k_ln<<<BB * LL, 256, 0, stream>>>(input, g, be, xln);
  k_qkv<<<(BB * LL / 16) * (N3 / 16) / 8, 256, 0, stream>>>(xln, wqkvT, bqkv, qk, vT);
  size_t shm = (size_t)(16 * LL + 256 + 16 + 8 * 256) * sizeof(float);  // ~140 KB
  k_attn<<<BB * HH * (LL / 16), 256, shm, stream>>>(qk, mask, vT, attn, outh);
  k_fc<<<(BB * LL / 16) * (DD / 16) / 8, 256, 0, stream>>>(outh, fcT, fcb, input, out);
}
